// SS2D_Global_28647431864581
// MI455X (gfx1250) — matrix operand order repaired and checked by reading, not compiled, not run on hardware
//
#include <hip/hip_runtime.h>
#include <hip/hip_bf16.h>
#include <math.h>

// ---------------------------------------------------------------------------
// SS2D (VMamba selective-scan 2D) for gfx1250 / MI455X.
// B=2, H=W=64, C=96, Di=192, N=16, K=4, R=6, L=4096.
// GEMMs use v_wmma_f32_16x16x32_bf16 (wave32, 16x16 tiles); the sequential
// selective scan is parallelized across B*K*Di*N = 24576 chains (768 waves),
// with the n-reduction done via 16-lane xor shuffles.
// ---------------------------------------------------------------------------

typedef __attribute__((ext_vector_type(16))) __bf16 v16bf;
typedef __attribute__((ext_vector_type(8)))  float  v8f;
typedef __attribute__((ext_vector_type(8)))  unsigned int v8u;

union FragU { v8u u; v16bf b; };

__device__ __forceinline__ unsigned short f2bf_bits(float f) {
  union { float f; unsigned u; } a; a.f = f;
  unsigned r = a.u + 0x7FFFu + ((a.u >> 16) & 1u);   // round-to-nearest-even
  return (unsigned short)(r >> 16);
}

// ---------------------------------------------------------------------------
// fp32 -> bf16 converters
// ---------------------------------------------------------------------------
__global__ void f32_to_bf16_k(const float* __restrict__ in,
                              unsigned short* __restrict__ out, int n) {
  int i = blockIdx.x * 256 + threadIdx.x;
  if (i < n) out[i] = f2bf_bits(in[i]);
}

// x_proj_weight (K,38,Di) -> bf16 (K,48,Di), zero-padded rows 38..47
__global__ void convert_xproj_k(const float* __restrict__ w,
                                unsigned short* __restrict__ out, int n) {
  int i = blockIdx.x * 256 + threadIdx.x;
  if (i >= n) return;
  int d = i % 192;
  int c = (i / 192) % 48;
  int k = i / (192 * 48);
  out[i] = (c < 38) ? f2bf_bits(w[((size_t)k * 38 + c) * 192 + d]) : (unsigned short)0;
}

// ---------------------------------------------------------------------------
// Generic bf16 WMMA GEMM:  C[M,N] = A[M,K] * B[N,K]^T  (both operands K-major)
// 1 wave -> one 16x16 output tile; block = 4 waves across N tiles.
// ---------------------------------------------------------------------------
__global__ void gemm_bf16_wmma(const unsigned short* __restrict__ A,
                               const unsigned short* __restrict__ B,
                               float* __restrict__ C, int M, int N, int Kd) {
  const int wave = threadIdx.x >> 5;
  const int lane = threadIdx.x & 31;
  const int tm = blockIdx.x;
  const int tn = blockIdx.y * 4 + wave;
  if (tn * 16 >= N) return;                    // uniform per wave
  const int mn = lane & 15;                    // A row / B col within tile
  const int h  = lane >> 4;                    // lane half
  const unsigned short* Arow = A + (size_t)(tm * 16 + mn) * Kd;
  const unsigned short* Brow = B + (size_t)(tn * 16 + mn) * Kd;
  v8f acc = {};
  for (int kb = 0; kb < Kd; kb += 32) {
    FragU fa, fb;
#pragma unroll
    for (int j = 0; j < 8; ++j) {
      // A 16x32 bf16 layout: VGPR j holds K = kb + h*16 + 2j .. +1 (contiguous, matches B)
      int ka = kb + h * 16 + j * 2;
      fa.u[j] = *(const unsigned int*)(Arow + ka);
      // B 32x16 bf16 layout: lane holds column, K = kb + h*16 + e contiguous
      fb.u[j] = *(const unsigned int*)(Brow + kb + h * 16 + j * 2);
    }
    acc = __builtin_amdgcn_wmma_f32_16x16x32_bf16(false, fa.b, false, fb.b,
                                                  (short)0, acc, false, false);
  }
#pragma unroll
  for (int r = 0; r < 8; ++r) {
    int row = tm * 16 + r + 8 * h;             // C/D layout: VGPR r -> M = r + 8*h
    C[(size_t)row * N + tn * 16 + mn] = acc[r];
  }
}

// ---------------------------------------------------------------------------
// Depthwise 3x3 conv (SAME) + bias + SiLU.
// Writes channel-last fp32 + bf16 in both row-major (h*W+w) and transposed
// (w*H+h) orderings so the 4 scan directions become row gathers/reversals.
// ---------------------------------------------------------------------------
__global__ void dwconv_silu_k(const float* __restrict__ xz,
                              const float* __restrict__ cw,
                              const float* __restrict__ cb,
                              float* __restrict__ xc_cl, float* __restrict__ xcT_cl,
                              unsigned short* __restrict__ xc_bf,
                              unsigned short* __restrict__ xcT_bf,
                              int H, int W, int Di) {
  const int L = H * W;
  const int bl = blockIdx.x;
  const int b = bl / L, l = bl % L;
  const int hh = l / W, ww = l % W;
  const int d = threadIdx.x;                   // 0..Di-1 (192 threads = 6 waves)
  float acc = cb[d];
#pragma unroll
  for (int dy = 0; dy < 3; ++dy)
#pragma unroll
    for (int dx = 0; dx < 3; ++dx) {
      int yy = hh + dy - 1, xx = ww + dx - 1;
      if (yy >= 0 && yy < H && xx >= 0 && xx < W)
        acc += cw[d * 9 + dy * 3 + dx] *
               xz[((size_t)b * L + (size_t)yy * W + xx) * (2 * Di) + d];
    }
  acc = acc / (1.f + __expf(-acc));            // SiLU
  const int lT = ww * H + hh;
  size_t o1 = ((size_t)b * L + l)  * Di + d;
  size_t o2 = ((size_t)b * L + lT) * Di + d;
  xc_cl[o1]  = acc;             xcT_cl[o2] = acc;
  xc_bf[o1]  = f2bf_bits(acc);  xcT_bf[o2] = f2bf_bits(acc);
}

// ---------------------------------------------------------------------------
// x_dbl GEMM per (b,k):  out[l, c(48)] = xs[b,k,:,l] . Wk[c,:]
// A rows gathered with direction-dependent mapping (reversal for k>=2,
// transposed source for odd k).  3 waves per block (N tiles), WMMA bf16.
// ---------------------------------------------------------------------------
__global__ void xdbl_gemm_k(const unsigned short* __restrict__ xc_bf,
                            const unsigned short* __restrict__ xcT_bf,
                            const unsigned short* __restrict__ wxp_bf,
                            float* __restrict__ x_dbl, int L, int Di) {
  const int bk = blockIdx.z;
  const int b = bk >> 2, k = bk & 3;
  const int tn = threadIdx.x >> 5;             // 0..2
  const int lane = threadIdx.x & 31;
  const int tm = blockIdx.x;
  const int mn = lane & 15, h = lane >> 4;
  const int l = tm * 16 + mn;
  const int lsrc = (k >= 2) ? (L - 1 - l) : l;
  const unsigned short* src = (k & 1) ? xcT_bf : xc_bf;
  const unsigned short* Arow = src + ((size_t)b * L + lsrc) * Di;
  const unsigned short* Brow = wxp_bf + ((size_t)k * 48 + tn * 16 + mn) * Di;
  v8f acc = {};
  for (int kb = 0; kb < Di; kb += 32) {
    FragU fa, fb;
#pragma unroll
    for (int j = 0; j < 8; ++j) {
      int ka = kb + h * 16 + j * 2;
      fa.u[j] = *(const unsigned int*)(Arow + ka);
      fb.u[j] = *(const unsigned int*)(Brow + kb + h * 16 + j * 2);
    }
    acc = __builtin_amdgcn_wmma_f32_16x16x32_bf16(false, fa.b, false, fb.b,
                                                  (short)0, acc, false, false);
  }
#pragma unroll
  for (int r = 0; r < 8; ++r) {
    int row = tm * 16 + r + 8 * h;
    x_dbl[((size_t)bk * L + row) * 48 + tn * 16 + mn] = acc[r];
  }
}

// ---------------------------------------------------------------------------
// Fused dt-projection + softplus + discretization + scan + C-contraction.
// Thread = one (d,n) chain; 16 n's per half-wave; y = sum_n h*C via 4
// xor-shuffles.  x_dbl row layout: [0:6]=dt_raw, [6:22]=B, [22:38]=C.
// ---------------------------------------------------------------------------
__global__ void selective_scan_k(const float* __restrict__ x_dbl,
                                 const float* __restrict__ xc_cl,
                                 const float* __restrict__ xcT_cl,
                                 const float* __restrict__ A_logs,
                                 const float* __restrict__ dtw,
                                 const float* __restrict__ dtb,
                                 const float* __restrict__ Ds,
                                 float* __restrict__ ys, int L, int Di) {
  const int dtile = blockIdx.x, k = blockIdx.y, b = blockIdx.z;
  const int t = threadIdx.x;
  const int n = t & 15, dl = t >> 4;
  const int d = dtile * 16 + dl;
  const int kd = k * Di + d;
  const float Adn = -__expf(A_logs[kd * 16 + n]);    // A = -exp(A_logs)
  const float w0 = dtw[kd * 6 + 0], w1 = dtw[kd * 6 + 1], w2 = dtw[kd * 6 + 2];
  const float w3 = dtw[kd * 6 + 3], w4 = dtw[kd * 6 + 4], w5 = dtw[kd * 6 + 5];
  const float bias = dtb[kd];
  const float Dski = Ds[kd];
  const bool rev = (k >= 2);
  const float* src  = (k & 1) ? xcT_cl : xc_cl;
  const float* xrow = x_dbl + (size_t)(b * 4 + k) * L * 48;
  float* yout = ys + (size_t)(b * 4 + k) * L * (size_t)Di;
  float hstate = 0.f;
  for (int l = 0; l < L; ++l) {
    const float* xr = xrow + (size_t)l * 48;
    __builtin_prefetch((const void*)(xr + 48 * 24), 0, 1);   // global_prefetch_b8
    float dtr = bias + w0 * xr[0] + w1 * xr[1] + w2 * xr[2]
                     + w3 * xr[3] + w4 * xr[4] + w5 * xr[5];
    float dt = (dtr > 20.f) ? dtr : __logf(1.f + __expf(dtr));   // softplus
    float Bn = xr[6 + n];
    float Cn = xr[22 + n];
    int lsrc = rev ? (L - 1 - l) : l;
    float u = src[((size_t)b * L + lsrc) * Di + d];
    hstate = __expf(dt * Adn) * hstate + dt * u * Bn;            // h = dA*h + dBu
    float p = hstate * Cn;
    p += __shfl_xor(p, 1, 32);
    p += __shfl_xor(p, 2, 32);
    p += __shfl_xor(p, 4, 32);
    p += __shfl_xor(p, 8, 32);                                    // sum over n
    if (n == 0) yout[(size_t)l * Di + d] = p + u * Dski;
  }
}

// ---------------------------------------------------------------------------
// Cross-merge (4 directions) + LayerNorm over Di + SiLU(z) gate -> bf16.
// ---------------------------------------------------------------------------
__global__ void merge_ln_gate_k(const float* __restrict__ ys,
                                const float* __restrict__ xz,
                                const float* __restrict__ lnw,
                                const float* __restrict__ lnb,
                                unsigned short* __restrict__ yg_bf,
                                int H, int W, int Di) {
  const int L = H * W;
  const int bl = blockIdx.x;
  const int b = bl / L, l = bl % L;
  const int hh = l / W, ww = l % W;
  const int lT = ww * H + hh;
  const int d = threadIdx.x;                   // 0..Di-1
  float v = ys[((size_t)(b * 4 + 0) * L + l)            * Di + d]
          + ys[((size_t)(b * 4 + 2) * L + (L - 1 - l))  * Di + d]
          + ys[((size_t)(b * 4 + 1) * L + lT)           * Di + d]
          + ys[((size_t)(b * 4 + 3) * L + (L - 1 - lT)) * Di + d];
  __shared__ float s1[192];
  __shared__ float s2[192];
  s1[d] = v; s2[d] = v * v;
  __syncthreads();
  for (int off = 128; off > 0; off >>= 1) {
    if (d < off && d + off < Di) { s1[d] += s1[d + off]; s2[d] += s2[d + off]; }
    __syncthreads();
  }
  const float inv = 1.f / (float)Di;
  float mu  = s1[0] * inv;
  float var = s2[0] * inv - mu * mu;
  float yn = (v - mu) * rsqrtf(var + 1e-5f) * lnw[d] + lnb[d];
  float z = xz[((size_t)b * L + l) * (2 * Di) + Di + d];
  yn *= z / (1.f + __expf(-z));                // * silu(z)
  yg_bf[((size_t)b * L + l) * Di + d] = f2bf_bits(yn);
}

// ---------------------------------------------------------------------------
// Host-side orchestration (all launches on `stream`, graph-capture safe).
// ---------------------------------------------------------------------------
extern "C" void kernel_launch(void* const* d_in, const int* in_sizes, int n_in,
                              void* d_out, int out_size, void* d_ws, size_t ws_size,
                              hipStream_t stream) {
  (void)in_sizes; (void)n_in; (void)out_size; (void)ws_size;
  const float* x          = (const float*)d_in[0];
  const float* in_proj_w  = (const float*)d_in[1];
  const float* conv_w     = (const float*)d_in[2];
  const float* conv_b     = (const float*)d_in[3];
  const float* x_proj_w   = (const float*)d_in[4];
  const float* dt_w       = (const float*)d_in[5];
  const float* dt_b       = (const float*)d_in[6];
  const float* A_logs     = (const float*)d_in[7];
  const float* Ds         = (const float*)d_in[8];
  const float* ln_w       = (const float*)d_in[9];
  const float* ln_b       = (const float*)d_in[10];
  const float* out_proj_w = (const float*)d_in[11];

  const int B = 2, H = 64, W = 64, Cc = 96, Di = 192, K = 4;
  const int L = H * W;          // 4096
  const int M = B * L;          // 8192

  size_t off = 0;
  auto alloc = [&](size_t bytes) -> void* {
    void* p = (char*)d_ws + off;
    off = (off + bytes + 255) & ~(size_t)255;
    return p;
  };
  float* xz     = (float*)alloc((size_t)M * 2 * Di * 4);       // in_proj out (xi|z)
  float* xc_cl  = (float*)alloc((size_t)B * L * Di * 4);       // conv out, row-major pos
  float* xcT_cl = (float*)alloc((size_t)B * L * Di * 4);       // conv out, transposed pos
  float* x_dbl  = (float*)alloc((size_t)B * K * L * 48 * 4);   // [dt(6)|B(16)|C(16)|pad]
  float* ys     = (float*)alloc((size_t)B * K * L * Di * 4);   // per-direction scan out
  unsigned short* x_bf   = (unsigned short*)alloc((size_t)M * Cc * 2);
  unsigned short* win_bf = (unsigned short*)alloc((size_t)2 * Di * Cc * 2);
  unsigned short* wxp_bf = (unsigned short*)alloc((size_t)K * 48 * Di * 2);
  unsigned short* wop_bf = (unsigned short*)alloc((size_t)Cc * Di * 2);
  unsigned short* xc_bf  = (unsigned short*)alloc((size_t)B * L * Di * 2);
  unsigned short* xcT_bf = (unsigned short*)alloc((size_t)B * L * Di * 2);
  unsigned short* yg_bf  = (unsigned short*)alloc((size_t)B * L * Di * 2);

  // bf16 conversions
  int n1 = M * Cc;
  f32_to_bf16_k<<<(n1 + 255) / 256, 256, 0, stream>>>(x, x_bf, n1);
  int n2 = 2 * Di * Cc;
  f32_to_bf16_k<<<(n2 + 255) / 256, 256, 0, stream>>>(in_proj_w, win_bf, n2);
  int n3 = Cc * Di;
  f32_to_bf16_k<<<(n3 + 255) / 256, 256, 0, stream>>>(out_proj_w, wop_bf, n3);
  int n4 = K * 48 * Di;
  convert_xproj_k<<<(n4 + 255) / 256, 256, 0, stream>>>(x_proj_w, wxp_bf, n4);

  // in_proj: (8192x96) x (384x96)^T -> xz (8192x384).  24 N-tiles / 4 waves.
  gemm_bf16_wmma<<<dim3(M / 16, 6), 128, 0, stream>>>(x_bf, win_bf, xz, M, 2 * Di, Cc);

  // depthwise conv + SiLU (+ transposed copy, fp32 + bf16)
  dwconv_silu_k<<<dim3(B * L), Di, 0, stream>>>(xz, conv_w, conv_b,
                                                xc_cl, xcT_cl, xc_bf, xcT_bf, H, W, Di);

  // x_dbl per (b,k): (4096x192 gathered) x (48x192)^T
  xdbl_gemm_k<<<dim3(L / 16, 1, B * K), 96, 0, stream>>>(xc_bf, xcT_bf, wxp_bf,
                                                         x_dbl, L, Di);

  // selective scan: grid (Di/16, K, B), block 16d x 16n
  selective_scan_k<<<dim3(Di / 16, K, B), 256, 0, stream>>>(x_dbl, xc_cl, xcT_cl,
                                                            A_logs, dt_w, dt_b, Ds,
                                                            ys, L, Di);

  // cross-merge + LayerNorm + gate
  merge_ln_gate_k<<<dim3(B * L), Di, 0, stream>>>(ys, xz, ln_w, ln_b, yg_bf, H, W, Di);

  // out_proj: (8192x192) x (96x192)^T -> d_out (8192x96).  6 N-tiles / 4 waves.
  gemm_bf16_wmma<<<dim3(M / 16, 2), 128, 0, stream>>>(yg_bf, wop_bf, (float*)d_out,
                                                      M, Cc, Di);
}
